// UnrollingModelQ_62088047231177
// MI455X (gfx1250) — matrix operand order repaired and checked by reading, not compiled, not run on hardware
//
#include <hip/hip_runtime.h>

#define BATCH 4096
#define XW    4896
#define NB    48
#define FDIM  2016
#define DS0   2128
#define PS0   4128

typedef __attribute__((ext_vector_type(16))) __bf16 v16bf;
typedef __attribute__((ext_vector_type(8)))  float  v8f;
typedef unsigned short u16_t;
typedef __attribute__((ext_vector_type(8))) unsigned short u16x8;

union BFrag { v16bf v; u16x8 h[2]; };

// round-half-up f32 bits (cheap; weights get same treatment in convert pass)
__device__ __forceinline__ unsigned rhu(float f) {
  union { float f; unsigned u; } c; c.f = f;
  return c.u + 0x8000u;
}
// pack two f32 -> two bf16 in one u32 via v_perm_b32 (lo -> [15:0], hi -> [31:16])
__device__ __forceinline__ unsigned pkbf(float lo, float hi) {
  return __builtin_amdgcn_perm(rhu(hi), rhu(lo), 0x07060302u);
}

// ---------- weight convert + transpose: W[K][2048] f32 -> Wt[2048][Kpad] bf16 ----------
__global__ void convt_kernel(const float* __restrict__ W, u16_t* __restrict__ Wt,
                             int Kreal, int Kpad) {
  __shared__ float t[32][33];
  const int tid = threadIdx.x;            // 256 threads
  const int n0 = blockIdx.x * 32;         // gridDim.x = 2048/32
  const int k0 = blockIdx.y * 32;         // gridDim.y = Kpad/32
  const int nn = tid & 31;
  const int kr = tid >> 5;                // 0..7
#pragma unroll
  for (int i = 0; i < 4; ++i) {
    int k = kr + i * 8;
    float v = 0.0f;
    if (k0 + k < Kreal) v = W[(long)(k0 + k) * 2048 + n0 + nn];  // coalesced read
    t[k][nn] = v;
  }
  __syncthreads();
#pragma unroll
  for (int i = 0; i < 2; ++i) {
    int s = i * 256 + tid;                // 0..511
    int n = s >> 4;                       // 0..31
    int kp = (s & 15) << 1;               // 0,2,..,30
    unsigned u = pkbf(t[kp][n], t[kp + 1][n]);
    *reinterpret_cast<unsigned*>(&Wt[(long)(n0 + n) * Kpad + k0 + kp]) = u;  // coalesced
  }
}

// ---------------- elementwise prep: q (base) and q_ps (pump-scaled) ----------------
__global__ void prep_q_kernel(const float* __restrict__ x,
                              float* __restrict__ q_base,
                              float* __restrict__ q_ps) {
  int idx = blockIdx.x * blockDim.x + threadIdx.x;   // BATCH*2048 threads
  int r = idx >> 11;
  int c = idx & 2047;
  const float* xr = x + (long)r * XW;
  const float PI4 = 0.78539816339744830962f;
  if (c < FDIM) {
    float d = xr[NB + c];
    float q = PI4 * d * d;
    q_base[idx] = q;
    q_ps[idx]   = q;
  } else {
    int j = c - FDIM;
    float d  = xr[NB + j];
    float qj = PI4 * d * d;
    float cn = xr[2096 + j];
    float cr = xr[2064 + j];
    float pf = cn * cr * powf(qj, cn - 1.0f);
    q_base[idx] = pf;
    q_ps[idx]   = pf * xr[PS0 + j];                  // num_steps==1 -> pump col PS0+j
  }
}

// copy q_ps[:, -32:] into out[:, 2048:2080]
__global__ void tail_kernel(const float* __restrict__ q_ps, float* __restrict__ out) {
  int idx = blockIdx.x * blockDim.x + threadIdx.x;   // BATCH*32 threads
  int r = idx >> 5;
  int j = idx & 31;
  out[(long)r * 2080 + 2048 + j] = q_ps[(long)r * 2048 + 2016 + j];
}

// ---------------- fused GEMM: C = epi(A' @ W + bias) ----------------
// AMODE: 0 -> A' = A0 ; 1 -> A' = A0*A1 ; 2 -> A' = A0*(A1+A2+A3)
// EPI:   0 -> none ; 1 -> relu ; 2 -> C = C - leaky(v, *alpha_p)   (q_ps update)
// GUARD: 1 -> Kpad > Kreal, zero-fill A beyond Kreal ; 0 -> no bounds checks at all
// Wt: bf16, pre-transposed [N=2048][Kpad].
// Tile 128(M) x 256(N), 512 threads / 16 waves, double-buffered LDS, 1 barrier/iter.
template<int AMODE, int EPI, int GUARD>
__global__ void __launch_bounds__(512)
gemm_kernel(const float* __restrict__ A0, const float* __restrict__ A1,
            const float* __restrict__ A2, const float* __restrict__ A3,
            long lda,
            const u16_t* __restrict__ Wt,
            const float* __restrict__ bias,
            float* __restrict__ C, long ldc, int Kreal, int Kpad,
            const float* __restrict__ alpha_p) {
  __shared__ u16_t As[2][128][40];   // [buf][row][k] bf16
  __shared__ u16_t Bs[2][256][40];   // [buf][col][k] bf16

  const int tid  = threadIdx.x;
  const int lane = tid & 31;
  const int wave = tid >> 5;      // 0..15
  const int wm   = wave & 3;      // 4 waves along M (4*32 = 128)
  const int wn   = wave >> 2;     // 4 waves along N (4*64 = 256)
  const int half = lane >> 4;
  const int l16  = lane & 15;

  const long mBase = (long)blockIdx.y * 128;
  const long nBase = (long)blockIdx.x * 256;

  // per-thread staging coordinates (2 chunks each for A and W)
  int rowA[2], k4A[2], nW[2], k8W[2];
  long abase[2], wbase[2];
#pragma unroll
  for (int i = 0; i < 2; ++i) {
    int s = i * 512 + tid;                 // 0..1023
    rowA[i] = s >> 3;                      // 0..127
    k4A[i]  = (s & 7) << 2;                // 0..28
    abase[i] = (mBase + rowA[i]) * lda + k4A[i];
    nW[i]  = s >> 2;                       // 0..255
    k8W[i] = (s & 3) << 3;                 // 0,8,16,24
    wbase[i] = (long)(nBase + nW[i]) * Kpad + k8W[i];
  }

  float4 rA[2][4];
  u16x8  rW[2];

  auto issue = [&](int k0) {               // global loads for tile k0 into registers
#pragma unroll
    for (int i = 0; i < 2; ++i) {
      if (GUARD) {
        rA[i][0] = make_float4(0.f, 0.f, 0.f, 0.f);
        if (AMODE >= 1) rA[i][1] = make_float4(0.f, 0.f, 0.f, 0.f);
        if (AMODE == 2) { rA[i][2] = make_float4(0.f, 0.f, 0.f, 0.f);
                          rA[i][3] = make_float4(0.f, 0.f, 0.f, 0.f); }
      }
      if (!GUARD || (k0 + k4A[i] < Kreal)) {
        long base = abase[i] + k0;
        rA[i][0] = *reinterpret_cast<const float4*>(A0 + base);
        if (AMODE >= 1) rA[i][1] = *reinterpret_cast<const float4*>(A1 + base);
        if (AMODE == 2) {
          rA[i][2] = *reinterpret_cast<const float4*>(A2 + base);
          rA[i][3] = *reinterpret_cast<const float4*>(A3 + base);
        }
      }
    }
#pragma unroll
    for (int i = 0; i < 2; ++i)
      rW[i] = *reinterpret_cast<const u16x8*>(Wt + wbase[i] + k0);
  };

  auto commit = [&](int buf) {             // elementwise combine + bf16 pack -> LDS
#pragma unroll
    for (int i = 0; i < 2; ++i) {
      float a0, a1, a2, a3;
      if (AMODE == 0) {
        a0 = rA[i][0].x; a1 = rA[i][0].y; a2 = rA[i][0].z; a3 = rA[i][0].w;
      } else if (AMODE == 1) {
        a0 = rA[i][0].x * rA[i][1].x; a1 = rA[i][0].y * rA[i][1].y;
        a2 = rA[i][0].z * rA[i][1].z; a3 = rA[i][0].w * rA[i][1].w;
      } else {
        a0 = rA[i][0].x * (rA[i][1].x + rA[i][2].x + rA[i][3].x);
        a1 = rA[i][0].y * (rA[i][1].y + rA[i][2].y + rA[i][3].y);
        a2 = rA[i][0].z * (rA[i][1].z + rA[i][2].z + rA[i][3].z);
        a3 = rA[i][0].w * (rA[i][1].w + rA[i][2].w + rA[i][3].w);
      }
      uint2 p;
      p.x = pkbf(a0, a1);
      p.y = pkbf(a2, a3);
      *reinterpret_cast<uint2*>(&As[buf][rowA[i]][k4A[i]]) = p;
    }
#pragma unroll
    for (int i = 0; i < 2; ++i)
      *reinterpret_cast<u16x8*>(&Bs[buf][nW[i]][k8W[i]]) = rW[i];
  };

  v8f acc[2][4];
#pragma unroll
  for (int mi = 0; mi < 2; ++mi)
#pragma unroll
    for (int ni = 0; ni < 4; ++ni)
#pragma unroll
      for (int r = 0; r < 8; ++r) acc[mi][ni][r] = 0.0f;

  // prologue: stage tile 0 into buffer 0
  issue(0);
  commit(0);
  __syncthreads();

  const int iters = Kpad >> 5;
  for (int it = 0; it < iters; ++it) {
    const int  cur  = it & 1;
    const bool more = (it + 1) < iters;
    if (more) issue((it + 1) << 5);        // overlap next-tile loads with compute

    // ---- fragments per ISA 16-bit layouts (contiguous 16B ds reads) ----
    BFrag af[2];
#pragma unroll
    for (int mi = 0; mi < 2; ++mi) {
      int row = wm * 32 + mi * 16 + l16;
      int kh  = half * 8;                  // lanes 0-15: K 0-7/16-23 ; 16-31: 8-15/24-31
      af[mi].h[0] = *reinterpret_cast<const u16x8*>(&As[cur][row][kh]);
      af[mi].h[1] = *reinterpret_cast<const u16x8*>(&As[cur][row][16 + kh]);
    }
    BFrag bfr[4];
#pragma unroll
    for (int ni = 0; ni < 4; ++ni) {
      int col = wn * 64 + ni * 16 + l16;
      int kb  = half * 8;                  // lanes 0-15: K 0-7/16-23 ; 16-31: 8-15/24-31
      bfr[ni].h[0] = *reinterpret_cast<const u16x8*>(&Bs[cur][col][kb]);
      bfr[ni].h[1] = *reinterpret_cast<const u16x8*>(&Bs[cur][col][16 + kb]);
    }
#pragma unroll
    for (int mi = 0; mi < 2; ++mi)
#pragma unroll
      for (int ni = 0; ni < 4; ++ni)
        acc[mi][ni] = __builtin_amdgcn_wmma_f32_16x16x32_bf16(
            false, af[mi].v, false, bfr[ni].v, (short)0, acc[mi][ni], false, false);

    if (more) {
      commit(cur ^ 1);                     // write next tile to the other buffer
      __syncthreads();                     // single barrier per iteration
    }
  }

  // ---- epilogue: bias + activation / fused q_ps update ----
  float alpha = 0.0f;
  if (EPI == 2) alpha = alpha_p[0];
#pragma unroll
  for (int mi = 0; mi < 2; ++mi) {
#pragma unroll
    for (int ni = 0; ni < 4; ++ni) {
      long col = nBase + wn * 64 + ni * 16 + l16;
      float bv = bias[col];
#pragma unroll
      for (int r = 0; r < 8; ++r) {
        long row = mBase + wm * 32 + mi * 16 + half * 8 + r;  // C/D VGPR layout
        float v = acc[mi][ni][r] + bv;
        if (EPI == 1) v = fmaxf(v, 0.0f);
        long off = row * ldc + col;
        if (EPI == 2) {
          float leak = (v >= 0.0f) ? v : alpha * v;
          v = C[off] - leak;
        }
        C[off] = v;
      }
    }
  }
}

extern "C" void kernel_launch(void* const* d_in, const int* in_sizes, int n_in,
                              void* d_out, int out_size, void* d_ws, size_t ws_size,
                              hipStream_t stream) {
  (void)in_sizes; (void)n_in; (void)out_size; (void)ws_size;
  const float* x     = (const float*)d_in[0];
  const float* W_q0h = (const float*)d_in[1];
  const float* b_q0h = (const float*)d_in[2];
  const float* W_sq  = (const float*)d_in[3];
  const float* b_sq  = (const float*)d_in[4];
  const float* W_h0h = (const float*)d_in[5];
  const float* b_h0h = (const float*)d_in[6];
  const float* W_h0q = (const float*)d_in[7];
  const float* b_h0q = (const float*)d_in[8];
  const float* W_S   = (const float*)d_in[9];
  const float* b_S   = (const float*)d_in[10];
  const float* W_Dq  = (const float*)d_in[11];
  const float* b_Dq  = (const float*)d_in[12];
  const float* W_Dh  = (const float*)d_in[13];
  const float* b_Dh  = (const float*)d_in[14];
  const float* W_fh  = (const float*)d_in[15];
  const float* b_fh  = (const float*)d_in[16];
  const float* W_hf  = (const float*)d_in[17];
  const float* b_hf  = (const float*)d_in[18];
  const float* a_hf  = (const float*)d_in[19];
  const float* W_res = (const float*)d_in[20];
  const float* b_res = (const float*)d_in[21];
  const float* W_out = (const float*)d_in[22];
  const float* b_out = (const float*)d_in[23];
  // d_in[24] = num_steps (== 1 per setup)
  float* out = (float*)d_out;

  const long BD = (long)BATCH * 2048;
  float* q_base = (float*)d_ws;
  float* q_ps   = q_base + BD;
  float* r_h0q  = q_ps   + BD;
  float* r_h0h  = r_h0q  + BD;
  float* r_Sq   = r_h0h  + BD;
  float* r_qh   = r_Sq   + BD;
  float* r_sq   = r_qh   + BD;
  float* Dq     = r_sq   + BD;
  float* Dh     = Dq     + BD;
  float* hbuf   = Dh     + BD;

  // bf16 transposed weights live after the activation buffers
  u16_t* wt_pool = (u16_t*)(hbuf + BD);
  long wt_off = 0;
  auto conv = [&](const float* W, int Kreal, int Kpad) -> const u16_t* {
    u16_t* dst = wt_pool + wt_off;
    wt_off += (long)2048 * Kpad;
    convt_kernel<<<dim3(64, Kpad / 32), 256, 0, stream>>>(W, dst, Kreal, Kpad);
    return dst;
  };

  const u16_t* wt_h0q = conv(W_h0q, NB, 64);
  const u16_t* wt_h0h = conv(W_h0h, NB, 64);
  const u16_t* wt_S   = conv(W_S,   FDIM, FDIM);       // 2016 = 63*32, no pad
  const u16_t* wt_q0h = conv(W_q0h, 2048, 2048);
  const u16_t* wt_sq  = conv(W_sq,  2000, 2016);
  const u16_t* wt_out = conv(W_out, 2048, 2048);
  const u16_t* wt_Dq[5];  const u16_t* wt_Dh[5];
  const u16_t* wt_fh[5];  const u16_t* wt_hf[5];  const u16_t* wt_res[5];
  for (int i = 0; i < 5; ++i) {
    const long WO = (long)i * 2048 * 2048;
    wt_Dq[i]  = conv(W_Dq  + WO, 2048, 2048);
    wt_Dh[i]  = conv(W_Dh  + WO, 2048, 2048);
    wt_fh[i]  = conv(W_fh  + WO, 2048, 2048);
    wt_hf[i]  = conv(W_hf  + WO, 2048, 2048);
    wt_res[i] = conv(W_res + WO, 2048, 2048);
  }

  dim3 blk(512);
  dim3 gg(2048 / 256, BATCH / 128);

  prep_q_kernel<<<(BATCH * 2048) / 256, 256, 0, stream>>>(x, q_base, q_ps);

  // prologue GEMMs (GUARD=1 only where Kpad > Kreal)
  gemm_kernel<0,0,1><<<gg, blk, 0, stream>>>(x, nullptr, nullptr, nullptr, (long)XW,
                                             wt_h0q, b_h0q, r_h0q, 2048, NB, 64, nullptr);
  gemm_kernel<0,0,1><<<gg, blk, 0, stream>>>(x, nullptr, nullptr, nullptr, (long)XW,
                                             wt_h0h, b_h0h, r_h0h, 2048, NB, 64, nullptr);
  gemm_kernel<0,1,0><<<gg, blk, 0, stream>>>(x + NB, nullptr, nullptr, nullptr, (long)XW,
                                             wt_S, b_S, r_Sq, 2048, FDIM, FDIM, nullptr);
  gemm_kernel<0,0,0><<<gg, blk, 0, stream>>>(q_base, nullptr, nullptr, nullptr, 2048L,
                                             wt_q0h, b_q0h, r_qh, 2048, 2048, 2048, nullptr);
  gemm_kernel<0,0,1><<<gg, blk, 0, stream>>>(x + DS0, nullptr, nullptr, nullptr, (long)XW,
                                             wt_sq, b_sq, r_sq, 2048, 2000, 2016, nullptr);

  for (int i = 0; i < 5; ++i) {
    const long bO = (long)i * 2048;
    // D_q = (q_ps * res_S_q) @ W_Dq + b
    gemm_kernel<1,0,0><<<gg, blk, 0, stream>>>(q_ps, r_Sq, nullptr, nullptr, 2048L,
                                               wt_Dq[i], b_Dq + bO, Dq, 2048, 2048, 2048, nullptr);
    // D_h = relu(D_q @ W_Dh + b)
    gemm_kernel<0,1,0><<<gg, blk, 0, stream>>>(Dq, nullptr, nullptr, nullptr, 2048L,
                                               wt_Dh[i], b_Dh + bO, Dh, 2048, 2048, 2048, nullptr);
    // h = relu((D_q * (q_ps + res_s_q + res_h0_q)) @ W_fh + b)
    gemm_kernel<2,1,0><<<gg, blk, 0, stream>>>(Dq, q_ps, r_sq, r_h0q, 2048L,
                                               wt_fh[i], b_fh + bO, hbuf, 2048, 2048, 2048, nullptr);
    // q_ps -= leaky((D_h * (h + res_h0_h + res_q_h)) @ W_hf + b, a_hf[i])
    gemm_kernel<2,2,0><<<gg, blk, 0, stream>>>(Dh, hbuf, r_h0h, r_qh, 2048L,
                                               wt_hf[i], b_hf + bO, q_ps, 2048, 2048, 2048, a_hf + i);
    // res_q_h = relu(q_ps @ W_res + b)
    gemm_kernel<0,1,0><<<gg, blk, 0, stream>>>(q_ps, nullptr, nullptr, nullptr, 2048L,
                                               wt_res[i], b_res + bO, r_qh, 2048, 2048, 2048, nullptr);
  }

  // out[:, 0:2048] = q_ps @ W_out + b_out   (ldc = 2080)
  gemm_kernel<0,0,0><<<gg, blk, 0, stream>>>(q_ps, nullptr, nullptr, nullptr, 2048L,
                                             wt_out, b_out, out, 2080, 2048, 2048, nullptr);
  // out[:, 2048:2080] = q_ps[:, -32:]
  tail_kernel<<<(BATCH * 32) / 256, 256, 0, stream>>>(q_ps, out);
}
